// LogicLayer_48309792145456
// MI455X (gfx1250) — matrix operand order repaired and checked by reading, not compiled, not run on hardware
//
#include <hip/hip_runtime.h>

#define DEV __device__ __forceinline__

typedef __attribute__((ext_vector_type(16))) __bf16 bf16x16;
typedef __attribute__((ext_vector_type(2)))  __bf16 bf16x2;
typedef __attribute__((ext_vector_type(8)))  float  v8f;
typedef __attribute__((ext_vector_type(4)))  unsigned int u32x4;

struct U8 { u32x4 lo, hi; };   // 32 bytes, 16B aligned

// ---- constants -------------------------------------------------------------
#define BB   16
#define NN   128
#define CC   64
#define HH   256
#define OO   64

// bf16 transposed weight offsets inside d_ws (units: ushort)
enum : int {
  DIR0_W1T = 0,
  DIR0_W2T = DIR0_W1T + 64 * 256,
  RED0_W1T = DIR0_W2T + 256 * 64,
  RED0_W2T = RED0_W1T + 128 * 256,
  EXP1_W1T = RED0_W2T + 256 * 64,
  EXP1_W2T = EXP1_W1T + 64 * 256,
  DIR1_W1T = EXP1_W2T + 256 * 64,
  DIR1_W2T = DIR1_W1T + 64 * 256,
  RED1_W1T = DIR1_W2T + 256 * 64,
  RED1_W2T = RED1_W1T + 128 * 256,
  EXP2_W1T = RED1_W2T + 256 * 64,
  EXP2_W2T = EXP2_W1T + 128 * 256,
  DIR2_W1T = EXP2_W2T + 256 * 64,
  DIR2_W2T = DIR2_W1T + 128 * 256,
  WTS_END  = DIR2_W2T + 256 * 64,      // 294912 ushorts = 589824 bytes
};
#define R1_FOFF (WTS_END / 2)          // float offset of reduce1 buffer
#define R2_FOFF (R1_FOFF + BB * 2 * CC)

// ---- helpers ---------------------------------------------------------------
DEV unsigned short f2bf_bits(float f) {
  unsigned u = __builtin_bit_cast(unsigned, f);
  unsigned r = u + 0x7FFFu + ((u >> 16) & 1u);   // round-to-nearest-even
  return (unsigned short)(r >> 16);
}

// packed 2xfp32 -> 2xbf16 (low = a, high = b)
DEV unsigned pack_bf16(float a, float b) {
#if __has_builtin(__builtin_amdgcn_cvt_pk_bf16_f32)
  bf16x2 p = __builtin_amdgcn_cvt_pk_bf16_f32(a, b);
  return __builtin_bit_cast(unsigned, p);
#elif __has_builtin(__builtin_amdgcn_perm)
  // one v_perm_b32: high halves of both floats = truncating bf16 conversion
  return __builtin_amdgcn_perm(__builtin_bit_cast(unsigned, b),
                               __builtin_bit_cast(unsigned, a), 0x07060302u);
#else
  return (unsigned)f2bf_bits(a) | ((unsigned)f2bf_bits(b) << 16);
#endif
}

// A-matrix fragment (16xK row-major bf16 tile in LDS) for one 32-deep k-step.
// CDNA5 16-bit A layout: lane<16 -> M=lane, K = {kb..kb+7, kb+16..kb+23};
// lane>=16 -> M=lane-16, kb += 8.
DEV bf16x16 frag_A_lds(const unsigned short* buf, int ldk, int lane, int k0) {
  int row = lane & 15;
  int kb  = k0 + ((lane >> 4) << 3);
  U8 u;
  u.lo = *(const u32x4*)(buf + row * ldk + kb);
  u.hi = *(const u32x4*)(buf + row * ldk + kb + 16);
  return __builtin_bit_cast(bf16x16, u);
}

// B-matrix fragment from column-major bf16 weights in global memory.
// CDNA5 16-bit B layout: lane holds column N=lane&15; K = {kb..kb+7, kb+16..kb+23},
// kb = k0 + 8*(lane>=16) (column-major storage -> 2x b128).
DEV bf16x16 frag_B_global(const unsigned short* Wt, int ldk, int ncol, int k0, int lane) {
  int n  = ncol + (lane & 15);
  int kb = k0 + ((lane >> 4) << 3);
  const u32x4* p = (const u32x4*)(Wt + n * ldk + kb);
  U8 u; u.lo = p[0]; u.hi = p[2];
  return __builtin_bit_cast(bf16x16, u);
}

// One 16-row 2-layer MLP (kin -> 256 -> 64), accumulating scale*result into facc.
DEV void mlp16(const unsigned short* __restrict__ Abuf, int kin,
               const unsigned short* __restrict__ W1t, const float* __restrict__ b1,
               const unsigned short* __restrict__ W2t, const float* __restrict__ b2,
               unsigned short* __restrict__ Hbuf,
               const float* srow, v8f* facc, int lane) {
  const int nl = lane & 15;
  const int mb = (lane >> 4) << 3;
  // GEMM1: A(16 x kin) * W1t -> hidden, bias + relu, to LDS (bf16 row-major)
  for (int hc = 0; hc < HH / 16; ++hc) {
    v8f acc = {};
    for (int k0 = 0; k0 < kin; k0 += 32) {
      bf16x16 a = frag_A_lds(Abuf, kin, lane, k0);
      bf16x16 w = frag_B_global(W1t, kin, hc * 16, k0, lane);
      acc = __builtin_amdgcn_wmma_f32_16x16x32_bf16(false, a, false, w,
                                                    (short)0, acc, false, false);
    }
    float bias = b1[hc * 16 + nl];
    const int n = hc * 16 + nl;
#pragma unroll
    for (int r = 0; r < 8; r += 2) {
      float v0 = fmaxf(acc[r]     + bias, 0.0f);
      float v1 = fmaxf(acc[r + 1] + bias, 0.0f);
      unsigned p = pack_bf16(v0, v1);
      Hbuf[(mb + r)     * HH + n] = (unsigned short)(p & 0xFFFFu);
      Hbuf[(mb + r + 1) * HH + n] = (unsigned short)(p >> 16);
    }
  }
  __syncthreads();
  // GEMM2: H(16 x 256) * W2t -> out, bias in C, scale into facc
  for (int oc = 0; oc < OO / 16; ++oc) {
    float bias = b2[oc * 16 + nl];
    v8f acc;
#pragma unroll
    for (int r = 0; r < 8; ++r) acc[r] = bias;
    for (int k0 = 0; k0 < HH; k0 += 32) {
      bf16x16 a = frag_A_lds(Hbuf, HH, lane, k0);
      bf16x16 w = frag_B_global(W2t, HH, oc * 16, k0, lane);
      acc = __builtin_amdgcn_wmma_f32_16x16x32_bf16(false, a, false, w,
                                                    (short)0, acc, false, false);
    }
#pragma unroll
    for (int r = 0; r < 8; ++r) facc[oc][r] += srow[r] * acc[r];
  }
  __syncthreads();
}

DEV float sigmoidf(float x) { return 1.0f / (1.0f + __expf(-x)); }

// ---- prep kernels ----------------------------------------------------------
__global__ void transpose_bf16_kernel(const float* __restrict__ src,
                                      unsigned short* __restrict__ dst,
                                      int rows, int cols) {
  int idx = blockIdx.x * blockDim.x + threadIdx.x;
  if (idx >= rows * cols) return;
  int r = idx / cols, c = idx % cols;
  dst[c * rows + r] = f2bf_bits(src[idx]);   // [rows,cols] -> col-major [cols][rows]
}

__global__ void reduce1_kernel(const float* __restrict__ x1, float* __restrict__ r1) {
  int idx = blockIdx.x * blockDim.x + threadIdx.x;   // B*C
  if (idx >= BB * CC) return;
  int b = idx >> 6, c = idx & 63;
  float mx = -1e30f, mn = 1e30f;
  for (int n = 0; n < NN; ++n) {
    float v = x1[(b * NN + n) * CC + c];
    mx = fmaxf(mx, v);
    mn = fminf(mn, v);
  }
  r1[b * (2 * CC) + 2 * c]     = mx;
  r1[b * (2 * CC) + 2 * c + 1] = mn;
}

__global__ void reduce2_kernel(const float* __restrict__ x2, float* __restrict__ r2) {
  int idx = blockIdx.x * blockDim.x + threadIdx.x;   // B*N*C
  if (idx >= BB * NN * CC) return;
  int c = idx & 63;
  int i = (idx >> 6) & 127;
  int b = idx >> 13;
  float mx = 0.0f, mn = 1.0f;                 // diag -> 0 for max, 1 for min
  for (int j = 0; j < NN; ++j) {
    float v = x2[((b * NN + i) * NN + j) * CC + c];
    float ve = (j == i) ? 0.0f : v;
    float vf = (j == i) ? 1.0f : v;
    mx = fmaxf(mx, ve);
    mn = fminf(mn, vf);
  }
  r2[(b * NN + i) * (2 * CC) + 2 * c]     = mx;
  r2[(b * NN + i) * (2 * CC) + 2 * c + 1] = mn;
}

// ---- group 0 + group 1 (small, 129 tiles) ----------------------------------
__global__ __launch_bounds__(32) void group01_kernel(
    const float* __restrict__ x0, const float* __restrict__ x1,
    const float* __restrict__ r1, const float* __restrict__ r2,
    const int* __restrict__ actb, const unsigned short* __restrict__ wts,
    const float* b1_dir0, const float* b2_dir0,
    const float* b1_red0, const float* b2_red0,
    const float* b1_exp1, const float* b2_exp1,
    const float* b1_dir1, const float* b2_dir1,
    const float* b1_red1, const float* b2_red1,
    float* __restrict__ out0, float* __restrict__ out1) {
  __shared__ __align__(16) unsigned short Abuf[16 * 128];
  __shared__ __align__(16) unsigned short Hbuf[16 * 256];
  const int lane = threadIdx.x;
  const int nl = lane & 15;
  const int mb = (lane >> 4) << 3;
  v8f facc[4] = {};

  if (blockIdx.x == 0) {
    // ---- group 0: rows are the 16 batches ----
    float s0[8], s1[8];
    bool any0 = false;
#pragma unroll
    for (int r = 0; r < 8; ++r) {
      int bb = mb + r;
      s0[r] = (float)actb[bb * 7 + 0];
      s1[r] = (float)actb[bb * 7 + 1];
    }
    for (int bb = 0; bb < BB; ++bb)
      any0 = any0 || actb[bb * 7 + 0] || actb[bb * 7 + 1];

    for (int t = lane * 2; t < 16 * 64; t += 64)
      *(unsigned*)(Abuf + t) = pack_bf16(x0[t], x0[t + 1]);   // row=batch, k=c
    __syncthreads();
    mlp16(Abuf, 64, wts + DIR0_W1T, b1_dir0, wts + DIR0_W2T, b2_dir0, Hbuf, s0, facc, lane);

    for (int t = lane * 2; t < 16 * 128; t += 64)
      *(unsigned*)(Abuf + t) = pack_bf16(r1[t], r1[t + 1]);
    __syncthreads();
    mlp16(Abuf, 128, wts + RED0_W1T, b1_red0, wts + RED0_W2T, b2_red0, Hbuf, s1, facc, lane);

    for (int oc = 0; oc < 4; ++oc)
#pragma unroll
      for (int r = 0; r < 8; ++r) {
        float v = any0 ? sigmoidf(facc[oc][r]) : 0.0f;
        out0[(mb + r) * OO + oc * 16 + nl] = v;
      }
  } else {
    // ---- group 1: tile = (b, n0..n0+15) ----
    int tile = blockIdx.x - 1;
    int b  = tile >> 3;
    int n0 = (tile & 7) * 16;
    float a2 = (float)actb[b * 7 + 2];
    float a3 = (float)actb[b * 7 + 3];
    float a4 = (float)actb[b * 7 + 4];
    float s2[8], s3[8], s4[8];
#pragma unroll
    for (int r = 0; r < 8; ++r) { s2[r] = a2; s3[r] = a3; s4[r] = a4; }
    bool any1 = false;
    for (int bb = 0; bb < BB; ++bb)
      any1 = any1 || actb[bb * 7 + 2] || actb[bb * 7 + 3] || actb[bb * 7 + 4];

    // expander: every row = x0[b,:]
    for (int t = lane * 2; t < 16 * 64; t += 64) {
      int k = t & 63;
      *(unsigned*)(Abuf + t) = pack_bf16(x0[b * CC + k], x0[b * CC + k + 1]);
    }
    __syncthreads();
    mlp16(Abuf, 64, wts + EXP1_W1T, b1_exp1, wts + EXP1_W2T, b2_exp1, Hbuf, s2, facc, lane);

    // direct: row r -> x1[b, n0+r, :]
    for (int t = lane * 2; t < 16 * 64; t += 64) {
      int r = t >> 6, k = t & 63;
      const float* s = &x1[(b * NN + n0 + r) * CC + k];
      *(unsigned*)(Abuf + t) = pack_bf16(s[0], s[1]);
    }
    __syncthreads();
    mlp16(Abuf, 64, wts + DIR1_W1T, b1_dir1, wts + DIR1_W2T, b2_dir1, Hbuf, s3, facc, lane);

    // reducer: row r -> r2[b, n0+r, :2C]
    for (int t = lane * 2; t < 16 * 128; t += 64) {
      int r = t >> 7, k = t & 127;
      const float* s = &r2[(b * NN + n0 + r) * (2 * CC) + k];
      *(unsigned*)(Abuf + t) = pack_bf16(s[0], s[1]);
    }
    __syncthreads();
    mlp16(Abuf, 128, wts + RED1_W1T, b1_red1, wts + RED1_W2T, b2_red1, Hbuf, s4, facc, lane);

    for (int oc = 0; oc < 4; ++oc)
#pragma unroll
      for (int r = 0; r < 8; ++r) {
        float v = any1 ? sigmoidf(facc[oc][r]) : 0.0f;
        out1[(b * NN + n0 + mb + r) * OO + oc * 16 + nl] = v;
      }
  }
}

// ---- group 2 (dominant: 16384 tiles, 192 WMMA each) ------------------------
__global__ __launch_bounds__(32) void group2_kernel(
    const float* __restrict__ x1, const float* __restrict__ x2,
    const int* __restrict__ actb, const unsigned short* __restrict__ wts,
    const float* b1_exp2, const float* b2_exp2,
    const float* b1_dir2, const float* b2_dir2,
    float* __restrict__ out2) {
  __shared__ __align__(16) unsigned short Abuf[16 * 128];
  __shared__ __align__(16) unsigned short Hbuf[16 * 256];
  const int lane = threadIdx.x;
  const int nl = lane & 15;
  const int mb = (lane >> 4) << 3;

  int bid = blockIdx.x;
  int j0 = (bid & 7) * 16;
  int i  = (bid >> 3) & 127;
  int b  = bid >> 10;

  float a5 = (float)actb[b * 7 + 5];
  float a6 = (float)actb[b * 7 + 6];
  float s5[8], s6[8];
#pragma unroll
  for (int r = 0; r < 8; ++r) { s5[r] = a5; s6[r] = a6; }
  bool any2 = false;
  for (int bb = 0; bb < BB; ++bb)
    any2 = any2 || actb[bb * 7 + 5] || actb[bb * 7 + 6];

  v8f facc[4] = {};

  // expander MLP: row r channels = concat(x1[b,i,:], x1[b,j0+r,:])
  for (int t = lane * 2; t < 16 * 128; t += 64) {
    int r = t >> 7, k = t & 127;
    int j = j0 + r;
    const float* s = (k < 64) ? &x1[(b * NN + i) * CC + k]
                              : &x1[(b * NN + j) * CC + (k - 64)];
    *(unsigned*)(Abuf + t) = pack_bf16(s[0], s[1]);
  }
  __syncthreads();
  mlp16(Abuf, 128, wts + EXP2_W1T, b1_exp2, wts + EXP2_W2T, b2_exp2, Hbuf, s5, facc, lane);

  // direct MLP: row r channels = concat(x2[b,i,j,:], x2[b,j,i,:])
  for (int t = lane * 2; t < 16 * 128; t += 64) {
    int r = t >> 7, k = t & 127;
    int j = j0 + r;
    const float* s = (k < 64) ? &x2[((b * NN + i) * NN + j) * CC + k]
                              : &x2[((b * NN + j) * NN + i) * CC + (k - 64)];
    *(unsigned*)(Abuf + t) = pack_bf16(s[0], s[1]);
  }
  __syncthreads();
  mlp16(Abuf, 128, wts + DIR2_W1T, b1_dir2, wts + DIR2_W2T, b2_dir2, Hbuf, s6, facc, lane);

  for (int oc = 0; oc < 4; ++oc)
#pragma unroll
    for (int r = 0; r < 8; ++r) {
      float v = any2 ? sigmoidf(facc[oc][r]) : 0.0f;
      int j = j0 + mb + r;
      out2[((b * NN + i) * NN + j) * OO + oc * 16 + nl] = v;
    }
}

// ---- launch ----------------------------------------------------------------
extern "C" void kernel_launch(void* const* d_in, const int* in_sizes, int n_in,
                              void* d_out, int out_size, void* d_ws, size_t ws_size,
                              hipStream_t stream) {
  (void)in_sizes; (void)n_in; (void)out_size; (void)ws_size;
  const float* x0   = (const float*)d_in[0];
  const float* x1   = (const float*)d_in[1];
  const float* x2   = (const float*)d_in[2];
  const int*   actb = (const int*)d_in[3];
  // params flat order: for each mlp {W1, b1, W2, b2}:
  // 4..7 dir0 | 8..11 red0 | 12..15 exp1 | 16..19 dir1 | 20..23 red1 | 24..27 exp2 | 28..31 dir2

  unsigned short* wts = (unsigned short*)d_ws;
  float* fws = (float*)d_ws;
  float* r1 = fws + R1_FOFF;
  float* r2 = fws + R2_FOFF;

  auto tr = [&](int in_idx, int off, int rows, int cols) {
    int n = rows * cols;
    transpose_bf16_kernel<<<(n + 255) / 256, 256, 0, stream>>>(
        (const float*)d_in[in_idx], wts + off, rows, cols);
  };
  tr(4,  DIR0_W1T, 64, 256);  tr(6,  DIR0_W2T, 256, 64);
  tr(8,  RED0_W1T, 128, 256); tr(10, RED0_W2T, 256, 64);
  tr(12, EXP1_W1T, 64, 256);  tr(14, EXP1_W2T, 256, 64);
  tr(16, DIR1_W1T, 64, 256);  tr(18, DIR1_W2T, 256, 64);
  tr(20, RED1_W1T, 128, 256); tr(22, RED1_W2T, 256, 64);
  tr(24, EXP2_W1T, 128, 256); tr(26, EXP2_W2T, 256, 64);
  tr(28, DIR2_W1T, 128, 256); tr(30, DIR2_W2T, 256, 64);

  reduce1_kernel<<<(BB * CC + 255) / 256, 256, 0, stream>>>(x1, r1);
  reduce2_kernel<<<(BB * NN * CC + 255) / 256, 256, 0, stream>>>(x2, r2);

  float* out0 = (float*)d_out;
  float* out1 = out0 + BB * OO;
  float* out2 = out1 + BB * NN * OO;

  group01_kernel<<<1 + (BB * NN) / 16, 32, 0, stream>>>(
      x0, x1, r1, r2, actb, wts,
      (const float*)d_in[5],  (const float*)d_in[7],
      (const float*)d_in[9],  (const float*)d_in[11],
      (const float*)d_in[13], (const float*)d_in[15],
      (const float*)d_in[17], (const float*)d_in[19],
      (const float*)d_in[21], (const float*)d_in[23],
      out0, out1);

  group2_kernel<<<BB * NN * (NN / 16), 32, 0, stream>>>(
      x1, x2, actb, wts,
      (const float*)d_in[25], (const float*)d_in[27],
      (const float*)d_in[29], (const float*)d_in[31],
      out2);
}
